// MambaLanguageModel_71073118814513
// MI455X (gfx1250) — compile-verified, matrix operand order checked by reading, not run on hardware
//
#include <hip/hip_runtime.h>

typedef __bf16 bf16_t;
typedef __attribute__((ext_vector_type(16))) __bf16 v16bf;
typedef __attribute__((ext_vector_type(8)))  float  v8f;

#define VOCAB    32000
#define D_MODEL  256
#define D_STATE  64
#define D_CONV   4
#define D_INNER  1024
#define HEADDIM  64
#define NHEADS   16
#define CONV_DIM 1152        // D_INNER + 2*D_STATE
#define D_IN_PROJ 2192       // 2*D_INNER + 2*D_STATE + NHEADS
#define D_IN_PROJ_PAD 2240   // padded to multiple of 64 for guard-free WMMA tiles
#define B_SZ     2
#define SEQ      512
#define T_TOK    1024        // B_SZ*SEQ
#define N_LAYERS 2

#define WMMA_BF16(a, b, c) \
    __builtin_amdgcn_wmma_f32_16x16x32_bf16(false, (a), false, (b), (short)0, (c), false, false)

#if __has_builtin(__builtin_amdgcn_s_wait_asynccnt)
#define WAIT_ASYNC(n) __builtin_amdgcn_s_wait_asynccnt(n)
#else
#define WAIT_ASYNC(n) asm volatile("s_wait_asynccnt " #n ::: "memory")
#endif

// ---------------------------------------------------------------- utilities

__global__ __launch_bounds__(256)
void cvt_bf16_kernel(const float* __restrict__ src, bf16_t* __restrict__ dst, int n) {
    int i = blockIdx.x * 256 + threadIdx.x;
    if (i < n) dst[i] = (bf16_t)src[i];
}

// Win [layer][2192][256] f32 -> [layer][2240][256] bf16, zero pad rows 2192..2239
__global__ __launch_bounds__(256)
void cvt_win_pad_kernel(const float* __restrict__ src, bf16_t* __restrict__ dst) {
    int i = blockIdx.x * 256 + threadIdx.x;          // over N_LAYERS*2240*256
    if (i >= N_LAYERS * D_IN_PROJ_PAD * D_MODEL) return;
    int layer = i / (D_IN_PROJ_PAD * D_MODEL);
    int rem   = i % (D_IN_PROJ_PAD * D_MODEL);
    int r = rem / D_MODEL, c = rem % D_MODEL;
    dst[i] = (r < D_IN_PROJ)
           ? (bf16_t)src[((size_t)layer * D_IN_PROJ + r) * D_MODEL + c]
           : (bf16_t)0.f;
}

__global__ __launch_bounds__(256)
void embed_kernel(const int* __restrict__ x, const float* __restrict__ emb,
                  bf16_t* __restrict__ h) {
    int i = blockIdx.x * 256 + threadIdx.x;          // over T_TOK*D_MODEL
    if (i >= T_TOK * D_MODEL) return;
    int t = i / D_MODEL, d = i % D_MODEL;
    h[i] = (bf16_t)emb[(size_t)x[t] * D_MODEL + d];
}

// ---------------------------------------------------------------- WMMA GEMM
// C[M,N] = A[M,K] * W[N,K]^T (+bias). Preconditions: M%64==0, N%64==0, K%32==0.
// Block = 128 threads (4 waves). Block tile 64(M) x 64(N): wave w owns rows
// m0 = by*64 + w*16, all four 16-wide N subtiles (B fragments shared via L0).
// K double-buffered: next-step fragments load before current WMMAs issue.

__device__ inline void store_tile(v8f acc, float* outF, bf16_t* outB,
                                  const float* bias, int m0, int nb, int N, int lane) {
    int col  = nb + (lane & 15);
    int mrow = m0 + ((lane >> 4) << 3);
    float bv = bias ? bias[col] : 0.f;
    #pragma unroll
    for (int r = 0; r < 8; ++r) {
        float v = acc[r] + bv;
        size_t idx = (size_t)(mrow + r) * N + col;
        if (outF) outF[idx] = v;
        if (outB) outB[idx] = (bf16_t)v;
    }
}

__global__ __launch_bounds__(128)
void gemm_bf16_kernel(const bf16_t* __restrict__ A, const bf16_t* __restrict__ W,
                      const float* __restrict__ bias,
                      float* __restrict__ outF, bf16_t* __restrict__ outB,
                      int M, int N, int K) {
    const int lane = threadIdx.x & 31;
    const int wave = threadIdx.x >> 5;
    const int m0   = blockIdx.y * 64 + wave * 16;
    const int nb   = blockIdx.x * 64;
    const int row  = lane & 15;
    const int koff = (lane >> 4) << 4;   // lane halves: K 0..15 / 16..31 of the step

    const bf16_t* Ap = A + (size_t)(m0 + row) * K + koff;
    const bf16_t* Wp = W + (size_t)(nb + row) * K + koff;
    const size_t ws16 = (size_t)16 * K;

    v8f acc0 = {}, acc1 = {}, acc2 = {}, acc3 = {};

    v16bf a  = *(const v16bf*)(Ap);
    v16bf b0 = *(const v16bf*)(Wp);
    v16bf b1 = *(const v16bf*)(Wp + ws16);
    v16bf b2 = *(const v16bf*)(Wp + 2 * ws16);
    v16bf b3 = *(const v16bf*)(Wp + 3 * ws16);

    for (int k0 = 0; k0 + 32 < K; k0 += 32) {
        const int kn = k0 + 32;
        v16bf an  = *(const v16bf*)(Ap + kn);
        v16bf bn0 = *(const v16bf*)(Wp + kn);
        v16bf bn1 = *(const v16bf*)(Wp + ws16 + kn);
        v16bf bn2 = *(const v16bf*)(Wp + 2 * ws16 + kn);
        v16bf bn3 = *(const v16bf*)(Wp + 3 * ws16 + kn);
        if (kn + 64 <= K) __builtin_prefetch(Ap + kn + 64, 0, 1); // global_prefetch_b8
        acc0 = WMMA_BF16(a, b0, acc0);
        acc1 = WMMA_BF16(a, b1, acc1);
        acc2 = WMMA_BF16(a, b2, acc2);
        acc3 = WMMA_BF16(a, b3, acc3);
        a = an; b0 = bn0; b1 = bn1; b2 = bn2; b3 = bn3;
    }
    acc0 = WMMA_BF16(a, b0, acc0);
    acc1 = WMMA_BF16(a, b1, acc1);
    acc2 = WMMA_BF16(a, b2, acc2);
    acc3 = WMMA_BF16(a, b3, acc3);

    store_tile(acc0, outF, outB, bias, m0, nb,      N, lane);
    store_tile(acc1, outF, outB, bias, m0, nb + 16, N, lane);
    store_tile(acc2, outF, outB, bias, m0, nb + 32, N, lane);
    store_tile(acc3, outF, outB, bias, m0, nb + 48, N, lane);
}

// ------------------------------------------------- conv (depthwise, causal) + SiLU

__global__ __launch_bounds__(256)
void conv_silu_kernel(const float* __restrict__ zx, const float* __restrict__ convw,
                      const float* __restrict__ convb, float* __restrict__ xbc) {
    int i = blockIdx.x * 256 + threadIdx.x;          // over T_TOK*CONV_DIM
    if (i >= T_TOK * CONV_DIM) return;
    int t = i / CONV_DIM, c = i % CONV_DIM;
    int l = t % SEQ, b = t / SEQ;
    float acc = convb[c];
    #pragma unroll
    for (int j = 0; j < D_CONV; ++j) {
        int li = l - (D_CONV - 1) + j;
        if (li >= 0)
            acc += zx[(size_t)(b * SEQ + li) * D_IN_PROJ_PAD + D_INNER + c] * convw[c * D_CONV + j];
    }
    acc = acc / (1.f + __expf(-acc));                // silu
    xbc[i] = acc;
}

// --------------------------------------------------------- dt softplus + dA

__global__ __launch_bounds__(256)
void dt_kernel(const float* __restrict__ zx, const float* __restrict__ dt_bias,
               const float* __restrict__ A_log,
               float* __restrict__ dt_sp, float* __restrict__ dA) {
    int i = blockIdx.x * 256 + threadIdx.x;          // over T_TOK*NHEADS
    if (i >= T_TOK * NHEADS) return;
    int t = i / NHEADS, hd = i % NHEADS;
    float v = zx[(size_t)t * D_IN_PROJ_PAD + (D_INNER + CONV_DIM) + hd] + dt_bias[hd];
    float sp = (v > 20.f) ? v : log1pf(__expf(v));   // softplus
    float A  = -__expf(A_log[hd]);
    dt_sp[i] = sp;
    dA[i]    = __expf(sp * A);
}

// --------------------------------------------------------------- SSM scan
// 32 blocks = (batch, head); state [64 x 64] in registers: thread owns
// (p = tid>>2, 16-wide n slice). B/C (512 B/step) staged into LDS with
// CDNA5 async-to-LDS loads, double-buffered one step ahead (ASYNCcnt).

__global__ __launch_bounds__(256)
void scan_kernel(const float* __restrict__ xbc, const float* __restrict__ dt_sp,
                 const float* __restrict__ dA, float* __restrict__ ys) {
    int b = blockIdx.x >> 4, hd = blockIdx.x & 15;
    int tid = threadIdx.x, lane = tid & 31;
    int p  = tid >> 2;
    int n0 = (tid & 3) << 4;
    float s[16];
    #pragma unroll
    for (int j = 0; j < 16; ++j) s[j] = 0.f;

    __shared__ float BC[2][128];     // [buf][ B: 0..63 | C: 64..127 ]
    // low 32 bits of a generic pointer to LDS == LDS byte offset (aperture layout)
    const unsigned lds0 = (unsigned)(unsigned long long)&BC[0][0];

    // prologue: async-stage step 0 (B|C contiguous, 512B) into buffer 0; wave 0 only
    if (tid < 32) {
        unsigned long long g =
            (unsigned long long)(xbc + (size_t)(b * SEQ) * CONV_DIM + D_INNER + lane * 4);
        unsigned ld = lds0 + lane * 16;
        asm volatile("global_load_async_to_lds_b128 %0, %1, off" :: "v"(ld), "v"(g) : "memory");
    }

    for (int l = 0; l < SEQ; ++l) {
        int t = b * SEQ + l;
        const float* xrow = xbc + (size_t)t * CONV_DIM;
        if (tid < 32) {               // wave-uniform branch: wave 0 manages staging
            if (l + 1 < SEQ) {        // issue next step, then retire current (<=1 left)
                unsigned long long g =
                    (unsigned long long)(xrow + CONV_DIM + D_INNER + lane * 4);
                unsigned ld = lds0 + ((unsigned)((l + 1) & 1)) * 512u + lane * 16;
                asm volatile("global_load_async_to_lds_b128 %0, %1, off" :: "v"(ld), "v"(g) : "memory");
                WAIT_ASYNC(1);
            } else {
                WAIT_ASYNC(0);
            }
        }
        __syncthreads();              // publish staged buffer to all waves
        const float* Bs = &BC[l & 1][0];
        const float* Cs = &BC[l & 1][64];
        float sdA = dA[t * NHEADS + hd];
        float dxp = dt_sp[t * NHEADS + hd] * xrow[hd * HEADDIM + p];
        float acc = 0.f;
        #pragma unroll
        for (int j = 0; j < 16; ++j) {
            int n = n0 + j;
            s[j] = s[j] * sdA + dxp * Bs[n];
            acc += s[j] * Cs[n];
        }
        acc += __shfl_xor(acc, 1, 32);
        acc += __shfl_xor(acc, 2, 32);
        if ((tid & 3) == 0)
            ys[(size_t)t * D_INNER + hd * HEADDIM + p] = acc;
        __syncthreads();              // all reads done before buffer is overwritten
    }
}

// --------------------------------------- +D skip, z-gate (SiLU), RMSNorm, ->bf16

__global__ __launch_bounds__(256)
void gate_norm_kernel(const float* __restrict__ ys, const float* __restrict__ xbc,
                      const float* __restrict__ zx, const float* __restrict__ Dv,
                      const float* __restrict__ nw, bf16_t* __restrict__ yout) {
    int t = blockIdx.x;
    int tid = threadIdx.x, lane = tid & 31, wave = tid >> 5;
    float v[4]; float ss = 0.f;
    #pragma unroll
    for (int i = 0; i < 4; ++i) {
        int e  = tid * 4 + i;
        int hd = e >> 6;
        float y = ys[(size_t)t * D_INNER + e] + xbc[(size_t)t * CONV_DIM + e] * Dv[hd];
        float z = zx[(size_t)t * D_IN_PROJ_PAD + e];
        y *= z / (1.f + __expf(-z));
        v[i] = y;
        ss += y * y;
    }
    #pragma unroll
    for (int m = 1; m < 32; m <<= 1) ss += __shfl_xor(ss, m, 32);
    __shared__ float red[8];
    if (lane == 0) red[wave] = ss;
    __syncthreads();
    float tot = 0.f;
    #pragma unroll
    for (int w = 0; w < 8; ++w) tot += red[w];
    float r = rsqrtf(tot / (float)D_INNER + 1e-5f);
    #pragma unroll
    for (int i = 0; i < 4; ++i) {
        int e = tid * 4 + i;
        yout[(size_t)t * D_INNER + e] = (bf16_t)(v[i] * r * nw[e]);
    }
}

// ------------------------------------------------------------------ driver

extern "C" void kernel_launch(void* const* d_in, const int* in_sizes, int n_in,
                              void* d_out, int out_size, void* d_ws, size_t ws_size,
                              hipStream_t stream) {
    const int*   x       = (const int*)  d_in[0];
    const float* emb     = (const float*)d_in[1];
    const float* Win     = (const float*)d_in[2];
    const float* convw   = (const float*)d_in[3];
    const float* convb   = (const float*)d_in[4];
    const float* dt_bias = (const float*)d_in[5];
    const float* A_log   = (const float*)d_in[6];
    const float* Dvec    = (const float*)d_in[7];
    const float* norm_w  = (const float*)d_in[8];
    const float* Wout    = (const float*)d_in[9];
    const float* Wlm     = (const float*)d_in[10];
    const float* blm     = (const float*)d_in[11];
    float* logits = (float*)d_out;

    // workspace carve-out (256B aligned)
    size_t off = 0;
    char* base = (char*)d_ws;
    auto carve = [&](size_t bytes) -> char* {
        off = (off + 255) & ~(size_t)255;
        char* p = base + off; off += bytes; return p;
    };
    bf16_t* h_bf    = (bf16_t*)carve((size_t)T_TOK * D_MODEL * 2);
    bf16_t* Win_bf  = (bf16_t*)carve((size_t)N_LAYERS * D_IN_PROJ_PAD * D_MODEL * 2);
    bf16_t* Wout_bf = (bf16_t*)carve((size_t)N_LAYERS * D_MODEL * D_INNER * 2);
    bf16_t* Wlm_bf  = (bf16_t*)carve((size_t)VOCAB * D_MODEL * 2);
    float*  zx      = (float*) carve((size_t)T_TOK * D_IN_PROJ_PAD * 4);
    float*  xbc     = (float*) carve((size_t)T_TOK * CONV_DIM * 4);
    float*  dt_sp   = (float*) carve((size_t)T_TOK * NHEADS * 4);
    float*  dAb     = (float*) carve((size_t)T_TOK * NHEADS * 4);
    float*  ysb     = (float*) carve((size_t)T_TOK * D_INNER * 4);
    bf16_t* y_bf    = (bf16_t*)carve((size_t)T_TOK * D_INNER * 2);

    auto blocks = [](size_t n) { return (unsigned)((n + 255) / 256); };

    // one-time weight conversions to bf16 (Win zero-padded to 2240 rows)
    {   size_t n = (size_t)N_LAYERS * D_IN_PROJ_PAD * D_MODEL;
        cvt_win_pad_kernel<<<blocks(n), 256, 0, stream>>>(Win, Win_bf); }
    {   size_t n = (size_t)N_LAYERS * D_MODEL * D_INNER;
        cvt_bf16_kernel<<<blocks(n), 256, 0, stream>>>(Wout, Wout_bf, (int)n); }
    {   size_t n = (size_t)VOCAB * D_MODEL;
        cvt_bf16_kernel<<<blocks(n), 256, 0, stream>>>(Wlm, Wlm_bf, (int)n); }

    embed_kernel<<<blocks((size_t)T_TOK * D_MODEL), 256, 0, stream>>>(x, emb, h_bf);

    for (int layer = 0; layer < N_LAYERS; ++layer) {
        const bf16_t* Wi = Win_bf  + (size_t)layer * D_IN_PROJ_PAD * D_MODEL;
        const bf16_t* Wo = Wout_bf + (size_t)layer * D_MODEL * D_INNER;

        // in_proj: zx[1024,2240] = h[1024,256] x WinPad[2240,256]^T
        {   dim3 g(D_IN_PROJ_PAD / 64, T_TOK / 64);
            gemm_bf16_kernel<<<g, 128, 0, stream>>>(h_bf, Wi, nullptr, zx, nullptr,
                                                    T_TOK, D_IN_PROJ_PAD, D_MODEL); }
        conv_silu_kernel<<<blocks((size_t)T_TOK * CONV_DIM), 256, 0, stream>>>(
            zx, convw + (size_t)layer * CONV_DIM * D_CONV,
            convb + (size_t)layer * CONV_DIM, xbc);
        dt_kernel<<<blocks((size_t)T_TOK * NHEADS), 256, 0, stream>>>(
            zx, dt_bias + layer * NHEADS, A_log + layer * NHEADS, dt_sp, dAb);
        scan_kernel<<<B_SZ * NHEADS, 256, 0, stream>>>(xbc, dt_sp, dAb, ysb);
        gate_norm_kernel<<<T_TOK, 256, 0, stream>>>(
            ysb, xbc, zx, Dvec + layer * NHEADS, norm_w + (size_t)layer * D_INNER, y_bf);

        // out_proj: h[1024,256] = y[1024,1024] x Wout[256,1024]^T (bf16 out only)
        {   dim3 g(D_MODEL / 64, T_TOK / 64);
            gemm_bf16_kernel<<<g, 128, 0, stream>>>(y_bf, Wo, nullptr, nullptr, h_bf,
                                                    T_TOK, D_MODEL, D_INNER); }
    }

    // LM head: logits[1024,32000] = h[1024,256] x Wlm[32000,256]^T + blm
    {   dim3 g(VOCAB / 64, T_TOK / 64);
        gemm_bf16_kernel<<<g, 128, 0, stream>>>(h_bf, Wlm_bf, blm, logits, nullptr,
                                                T_TOK, VOCAB, D_MODEL); }
}
